// GRU_27719718928938
// MI455X (gfx1250) — matrix operand order repaired and checked by reading, not compiled, not run on hardware
//
#include <hip/hip_runtime.h>

// ---------------- problem constants ----------------
#define B_   64
#define S_   512
#define I_   512
#define H_   1024
#define LDK  2048              // comb row stride (max K)
#define NBLK 16                // persistent grid: 16 blocks = 2 M-groups x 8 N-groups
#define NTHR 256               // 8 waves (wave32) per block -> 128 waves total

typedef __attribute__((ext_vector_type(16))) __bf16 v16bf;
typedef __attribute__((ext_vector_type(8)))  float  v8f;
typedef __attribute__((ext_vector_type(4)))  unsigned int u32x4;
typedef __attribute__((ext_vector_type(8)))  int    i32x8;
typedef __attribute__((ext_vector_type(4)))  int    i32x4;

// ---------------- grid-wide sense barrier ----------------
__device__ __forceinline__ void gbar(int* cnt, int* gen) {
  __syncthreads();
  if (threadIdx.x == 0) {
    __threadfence();  // make my block's writes visible device-wide
    int g = __hip_atomic_load(gen, __ATOMIC_RELAXED, __HIP_MEMORY_SCOPE_AGENT);
    int t = __hip_atomic_fetch_add(cnt, 1, __ATOMIC_ACQ_REL, __HIP_MEMORY_SCOPE_AGENT);
    if (t == NBLK - 1) {
      __hip_atomic_store(cnt, 0, __ATOMIC_RELAXED, __HIP_MEMORY_SCOPE_AGENT);
      __hip_atomic_fetch_add(gen, 1, __ATOMIC_RELEASE, __HIP_MEMORY_SCOPE_AGENT);
    } else {
      while (__hip_atomic_load(gen, __ATOMIC_ACQUIRE, __HIP_MEMORY_SCOPE_AGENT) == g) {
        __builtin_amdgcn_s_sleep(1);
      }
    }
    __threadfence();
  }
  __syncthreads();
}

// Stage 32 rows (m0g..m0g+31) x K bf16 of comb into LDS.
// Preferred path: one Tensor-Data-Mover descriptor issued by wave 0
// (D# layout per CDNA5 ISA; 2D tile, packed rows in LDS).
// This toolchain exposes the 6-arg builtin: (g0, g1, g2, g3, extra_g, cpol).
__device__ __forceinline__ void stage_A(__bf16* sA, const __bf16* comb, int m0g, int K) {
#if __has_builtin(__builtin_amdgcn_tensor_load_to_lds)
  if ((threadIdx.x >> 5) == 0) {           // wave 0 issues the DMA
    unsigned long long ga =
        (unsigned long long)(uintptr_t)(comb + (size_t)m0g * LDK);
    unsigned lds_off = (unsigned)(uintptr_t)sA;   // low 32b of generic = LDS offset
    u32x4 g0;
    g0[0] = 1u;                                    // count=1 (valid), user mode
    g0[1] = lds_off;                               // lds_addr
    g0[2] = (unsigned)ga;                          // global_addr[31:0]
    g0[3] = (unsigned)((ga >> 32) & 0x01FFFFFFu)   // global_addr[56:32]
            | (2u << 30);                          // type=2 ("image")
    i32x8 g1 = {0, 0, 0, 0, 0, 0, 0, 0};
    g1[0] = 1 << 16;                               // data_size=1 -> 2 bytes
    g1[1] = (int)((LDK & 0xFFFF) << 16);           // tensor_dim0[15:0]
    g1[2] = (int)((LDK >> 16) | (B_ << 16));       // tensor_dim0[31:16] | tensor_dim1[15:0]
    g1[3] = (int)(K << 16);                        // tensor_dim1[31:16]=0 | tile_dim0=K
    g1[4] = 32;                                    // tile_dim1=32 rows, tile_dim2=0
    g1[5] = LDK;                                   // tensor_dim0_stride[31:0]
    g1[6] = 0;                                     // stride[47:32] | dim1_stride lo
    g1[7] = 0;
    i32x4 gz  = {0, 0, 0, 0};
    i32x8 gz8 = {0, 0, 0, 0, 0, 0, 0, 0};
    __builtin_amdgcn_tensor_load_to_lds(g0, g1, gz, gz, gz8, 0);
    __builtin_amdgcn_s_wait_tensorcnt(0);
  }
#else
  {
    const int rowChunks = K >> 3;                  // 8 bf16 (16B) per chunk
    const int total = 32 * rowChunks;
    for (int c = threadIdx.x; c < total; c += NTHR) {
      int row = c / rowChunks;
      int col = (c % rowChunks) << 3;
      *(u32x4*)(sA + row * K + col) =
          *(const u32x4*)(comb + (m0g + row) * LDK + col);
    }
  }
#endif
  __syncthreads();
}

// A fragment 16x32 bf16 (CDNA5 layout): lanes 0-15 row=lane K{0..7,16..23},
// lanes 16-31 row=lane-16 K{8..15,24..31}.
__device__ __forceinline__ v16bf load_afrag(const __bf16* sA, int K, int k0, int lane) {
  v16bf a;
  u32x4* ap = (u32x4*)&a;
  const __bf16* p = sA + (lane & 15) * K + k0 + ((lane >> 4) << 3);
  ap[0] = *(const u32x4*)p;          // k .. k+7
  ap[1] = *(const u32x4*)(p + 16);   // k+16 .. k+23
  return a;
}

// B fragment 32x16 bf16 = W^T tile: B[k][n] = W[n0+n][k0+k]; W row-major [H_, K].
__device__ __forceinline__ v16bf load_bfrag(const __bf16* W, int K, int n0, int k0, int lane) {
  v16bf b;
  u32x4* bp = (u32x4*)&b;
  const __bf16* p = W + (size_t)(n0 + (lane & 15)) * K + k0 + ((lane >> 4) << 3);
  bp[0] = *(const u32x4*)p;
  bp[1] = *(const u32x4*)(p + 16);
  return b;
}

__device__ __forceinline__ float sigmoidf_(float x) {
  return 1.0f / (1.0f + __expf(-x));
}

#define WMMA_BF16(A, Bf, C) \
  __builtin_amdgcn_wmma_f32_16x16x32_bf16(false, (A), false, (Bf), (short)0, (C), false, false)

// ---------------- prologue: weight fp32->bf16, h=0, barrier init ----------------
__global__ void gru_init_kernel(
    const float* Wz0f, const float* Wr0f, const float* Wn0f,
    const float* Wz1f, const float* Wr1f, const float* Wn1f,
    __bf16* wz0, __bf16* wr0, __bf16* wn0,
    __bf16* wz1, __bf16* wr1, __bf16* wn1,
    float* h0, float* h1, int* cnt, int* gen) {
  const int g = blockIdx.x * blockDim.x + threadIdx.x;
  const int stride = gridDim.x * blockDim.x;
  const int W0 = H_ * (I_ + H_);
  const int W1 = H_ * (H_ + H_);
  for (int i = g; i < W0; i += stride) {
    wz0[i] = (__bf16)Wz0f[i];
    wr0[i] = (__bf16)Wr0f[i];
    wn0[i] = (__bf16)Wn0f[i];
  }
  for (int i = g; i < W1; i += stride) {
    wz1[i] = (__bf16)Wz1f[i];
    wr1[i] = (__bf16)Wr1f[i];
    wn1[i] = (__bf16)Wn1f[i];
  }
  for (int i = g; i < B_ * H_; i += stride) { h0[i] = 0.0f; h1[i] = 0.0f; }
  if (g == 0) { *cnt = 0; *gen = 0; }
}

// ---------------- persistent GRU kernel ----------------
// 16 blocks = 2 M-groups (32 rows each) x 8 N-groups (8 N-tiles each).
// Wave w of a block owns N-tile (Ngroup*8 + w) and BOTH M-tiles of its M-group:
// every B fragment is reused across 2 WMMAs; z/r share every A fragment.
__global__ void __launch_bounds__(NTHR, 1) gru_persistent_kernel(
    const float* __restrict__ x,
    const float* __restrict__ bz0, const float* __restrict__ br0, const float* __restrict__ bn0,
    const float* __restrict__ bz1, const float* __restrict__ br1, const float* __restrict__ bn1,
    const __bf16* __restrict__ wz0, const __bf16* __restrict__ wr0, const __bf16* __restrict__ wn0,
    const __bf16* __restrict__ wz1, const __bf16* __restrict__ wr1, const __bf16* __restrict__ wn1,
    float* h0, float* h1, float* rbuf, __bf16* comb,
    float* out, int* cnt, int* gen) {
  __shared__ __bf16 sA[32 * LDK];   // 128 KB A-panel (32 rows x up-to-2048 K)

  const int tid   = threadIdx.x;
  const int lane  = tid & 31;
  const int wave  = tid >> 5;                       // 0..7
  const int mgrp  = blockIdx.x >> 3;                // 0..1
  const int ngrp  = blockIdx.x & 7;                 // 0..7
  const int m0g   = mgrp << 5;                      // 0 or 32 (block's 32 rows)
  const int n0    = ((ngrp << 3) + wave) << 4;      // this wave's N-tile: 0..1008
  const int gtid  = blockIdx.x * NTHR + tid;        // 0..4095

  // C/D lane coords
  const int nn  = n0 + (lane & 15);
  const int mbr = (lane >> 4) << 3;                 // +0 or +8 within a 16-row tile

  for (int t = 0; t < S_; ++t) {
    const float* inp_h = h0;  // layer-1 input (updated h0)
    for (int l = 0; l < 2; ++l) {
      const int Kin = (l == 0) ? I_ : H_;
      const int K   = Kin + H_;
      float* hbuf = (l == 0) ? h0 : h1;
      const __bf16* Wz = l ? wz1 : wz0;
      const __bf16* Wr = l ? wr1 : wr0;
      const __bf16* Wn = l ? wn1 : wn0;
      const float* Bz = l ? bz1 : bz0;
      const float* Br = l ? br1 : br0;
      const float* Bn = l ? bn1 : bn0;

      // ---- phase 1: comb = [bf16(inp), bf16(h)] ----
      for (int i = gtid; i < B_ * K; i += NBLK * NTHR) {
        int b = i / K, c = i % K;
        float v;
        if (c < Kin)
          v = (l == 0) ? x[(b * S_ + t) * I_ + c] : inp_h[b * H_ + c];
        else
          v = hbuf[b * H_ + (c - Kin)];
        comb[b * LDK + c] = (__bf16)v;
      }
      gbar(cnt, gen);

      // ---- phase 2: z & r GEMMs; A staged once via TDM, B reused over 2 M-tiles ----
      stage_A(sA, comb, m0g, K);
      v8f accz0 = {}, accz1 = {}, accr0 = {}, accr1 = {};
      const __bf16* sA1 = sA + 16 * K;              // second M-tile of the panel
      for (int k0 = 0; k0 < K; k0 += 32) {
        v16bf a0 = load_afrag(sA,  K, k0, lane);
        v16bf a1 = load_afrag(sA1, K, k0, lane);
        v16bf fz = load_bfrag(Wz, K, n0, k0, lane);
        v16bf fr = load_bfrag(Wr, K, n0, k0, lane);
        if (k0 + 32 < K) {   // stream weights ahead (global_prefetch_b8)
          __builtin_prefetch(Wz + (size_t)(n0 + (lane & 15)) * K + k0 + 32, 0, 1);
          __builtin_prefetch(Wr + (size_t)(n0 + (lane & 15)) * K + k0 + 32, 0, 1);
        }
        accz0 = WMMA_BF16(a0, fz, accz0);
        accz1 = WMMA_BF16(a1, fz, accz1);
        accr0 = WMMA_BF16(a0, fr, accr0);
        accr1 = WMMA_BF16(a1, fr, accr1);
      }
      {
        const float bzv = Bz[nn], brv = Br[nn];
        #pragma unroll
        for (int v = 0; v < 8; ++v) {
          accz0[v] = sigmoidf_(accz0[v] + bzv);     // keep z in registers
          accz1[v] = sigmoidf_(accz1[v] + bzv);
          rbuf[(m0g +      mbr + v) * H_ + nn] = sigmoidf_(accr0[v] + brv);
          rbuf[(m0g + 16 + mbr + v) * H_ + nn] = sigmoidf_(accr1[v] + brv);
        }
      }
      gbar(cnt, gen);

      // ---- phase 3: comb h-part <- bf16(r * h) ----
      for (int i = gtid; i < B_ * H_; i += NBLK * NTHR) {
        int b = i / H_, j = i % H_;
        comb[b * LDK + Kin + j] = (__bf16)(rbuf[i] * hbuf[i]);
      }
      gbar(cnt, gen);

      // ---- phase 4: n GEMM + fused h update ----
      stage_A(sA, comb, m0g, K);                    // r*h part changed
      v8f accn0 = {}, accn1 = {};
      for (int k0 = 0; k0 < K; k0 += 32) {
        v16bf a0 = load_afrag(sA,  K, k0, lane);
        v16bf a1 = load_afrag(sA1, K, k0, lane);
        v16bf fn = load_bfrag(Wn, K, n0, k0, lane);
        if (k0 + 32 < K)
          __builtin_prefetch(Wn + (size_t)(n0 + (lane & 15)) * K + k0 + 32, 0, 1);
        accn0 = WMMA_BF16(a0, fn, accn0);
        accn1 = WMMA_BF16(a1, fn, accn1);
      }
      {
        const float bnv = Bn[nn];
        #pragma unroll
        for (int mt = 0; mt < 2; ++mt) {
          const int mb = m0g + (mt << 4) + mbr;
          #pragma unroll
          for (int v = 0; v < 8; ++v) {
            const int idx = (mb + v) * H_ + nn;
            float nval = tanhf((mt ? accn1[v] : accn0[v]) + bnv);
            float z = mt ? accz1[v] : accz0[v];
            float hnew = (1.0f - z) * hbuf[idx] + z * nval;
            hbuf[idx] = hnew;
            if (l == 1) {
              out[((mb + v) * S_ + t) * H_ + nn] = hnew;       // outputs [B,S,H]
              if (t == S_ - 1) out[B_ * S_ * H_ + idx] = hnew; // h_last [B,H]
            }
          }
        }
      }
      gbar(cnt, gen);
    }
  }
}

// ---------------- host launcher ----------------
extern "C" void kernel_launch(void* const* d_in, const int* in_sizes, int n_in,
                              void* d_out, int out_size, void* d_ws, size_t ws_size,
                              hipStream_t stream) {
  const float* x   = (const float*)d_in[0];
  const float* Wz0 = (const float*)d_in[1];  const float* bz0 = (const float*)d_in[2];
  const float* Wr0 = (const float*)d_in[3];  const float* br0 = (const float*)d_in[4];
  const float* Wn0 = (const float*)d_in[5];  const float* bn0 = (const float*)d_in[6];
  const float* Wz1 = (const float*)d_in[7];  const float* bz1 = (const float*)d_in[8];
  const float* Wr1 = (const float*)d_in[9];  const float* br1 = (const float*)d_in[10];
  const float* Wn1 = (const float*)d_in[11]; const float* bn1 = (const float*)d_in[12];
  float* out = (float*)d_out;

  // workspace carve-up (all chunks 256B-aligned)
  char* ws = (char*)d_ws;
  size_t off = 0;
  int* cnt = (int*)(ws + off);
  int* gen = cnt + 1;
  off += 256;
  float* h0   = (float*)(ws + off); off += (size_t)B_ * H_ * sizeof(float);
  float* h1   = (float*)(ws + off); off += (size_t)B_ * H_ * sizeof(float);
  float* rbuf = (float*)(ws + off); off += (size_t)B_ * H_ * sizeof(float);
  __bf16* comb = (__bf16*)(ws + off); off += (size_t)B_ * LDK * sizeof(__bf16);
  const size_t w0e = (size_t)H_ * (I_ + H_);   // 1.5M elems
  const size_t w1e = (size_t)H_ * (H_ + H_);   // 2.0M elems
  __bf16* wz0 = (__bf16*)(ws + off); off += w0e * sizeof(__bf16);
  __bf16* wr0 = (__bf16*)(ws + off); off += w0e * sizeof(__bf16);
  __bf16* wn0 = (__bf16*)(ws + off); off += w0e * sizeof(__bf16);
  __bf16* wz1 = (__bf16*)(ws + off); off += w1e * sizeof(__bf16);
  __bf16* wr1 = (__bf16*)(ws + off); off += w1e * sizeof(__bf16);
  __bf16* wn1 = (__bf16*)(ws + off); off += w1e * sizeof(__bf16);

  gru_init_kernel<<<dim3(1024), dim3(256), 0, stream>>>(
      Wz0, Wr0, Wn0, Wz1, Wr1, Wn1,
      wz0, wr0, wn0, wz1, wr1, wn1, h0, h1, cnt, gen);

  gru_persistent_kernel<<<dim3(NBLK), dim3(NTHR), 0, stream>>>(
      x, bz0, br0, bn0, bz1, br1, bn1,
      wz0, wr0, wn0, wz1, wr1, wn1,
      h0, h1, rbuf, comb, out, cnt, gen);
}
